// PsiNN_84688165142618
// MI455X (gfx1250) — matrix operand order repaired and checked by reading, not compiled, not run on hardware
//
#include <hip/hip_runtime.h>

// ---------------- problem constants ----------------
#define BB 64
#define TT 2048
#define II 25
#define RR 512
#define OO 50
#define ALPHA 0.5f

#define TC   4      // timesteps per chunk
#define HST  520    // padded row stride (elems): 1040B % 256B = 16 -> conflict-free
#define NTW  4      // N-tiles per wave (32 tiles / 8 waves)

typedef __attribute__((ext_vector_type(16))) __bf16 v16bf;
typedef __attribute__((ext_vector_type(8)))  __bf16 v8bf;
typedef __attribute__((ext_vector_type(8)))  float  v8f;
typedef __attribute__((ext_vector_type(4))) unsigned int v4u;
typedef __attribute__((ext_vector_type(8))) int v8i;
typedef __attribute__((ext_vector_type(4))) int v4i;
typedef unsigned int       u32;
typedef unsigned long long u64;
typedef unsigned short     u16;

#if __has_builtin(__builtin_amdgcn_tanhf)
#define TANHF(v) __builtin_amdgcn_tanhf(v)
#else
#define TANHF(v) tanhf(v)
#endif

__device__ __forceinline__ u16 f2bf(float f) {
  u32 u = __float_as_uint(f);
  u32 r = u + 0x7fffu + ((u >> 16) & 1u);   // round-to-nearest-even
  return (u16)(r >> 16);
}

// A-fragment (16x32 bf16): lane holds row M=lane%16,
// cols kBase + half*16 + [0..15] (contiguous, matches B)
__device__ __forceinline__ v16bf load_a_frag(const u16* p) {
  v8bf lo = *(const v8bf*)(p);
  v8bf hi = *(const v8bf*)(p + 8);
  v16bf r;
#pragma unroll
  for (int j = 0; j < 8; ++j) { r[j] = lo[j]; r[j + 8] = hi[j]; }
  return r;
}

// B-fragment from 16B-aligned LDS (contiguous 32B per lane, two b128s)
__device__ __forceinline__ v16bf load_b_frag16(const u16* p) {
  v8bf lo = *(const v8bf*)(p);
  v8bf hi = *(const v8bf*)(p + 8);
  v16bf r;
#pragma unroll
  for (int j = 0; j < 8; ++j) { r[j] = lo[j]; r[j + 8] = hi[j]; }
  return r;
}

__device__ __forceinline__ v8f wmma_bf16(v16bf a, v16bf b, v8f c) {
  return __builtin_amdgcn_wmma_f32_16x16x32_bf16(false, a, false, b, (short)0, c,
                                                 false, false);
}

// ---- Tensor Data Mover: build D# groups per CDNA5 ISA ----
__device__ __forceinline__ void tdm_load_to_lds(
    u64 gaddr, u32 ldsaddr, u32 ds_code,                  // data_size: 0=1B..3=8B
    u32 td0, u32 td1, u32 tile0, u32 tile1, u32 tile2,    // tensor/tile dims
    u64 s0, u64 s1, u32 td2,                              // strides (elems), dim2 len
    u32 pad_en, u32 pad_int, u32 pad_amt)                 // LDS padding controls
{
  v4u g0;
  g0[0] = 1u;                                             // count=1, user mode
  g0[1] = ldsaddr;                                        // lds_addr [63:32]
  g0[2] = (u32)gaddr;                                     // global_addr [95:64]
  g0[3] = (u32)((gaddr >> 32) & 0x1FFFFFFu) | (2u << 30); // addr[56:32] | type=2
  v8i g1;
  g1[0] = (int)((ds_code << 16) | (pad_en << 20) | (pad_int << 22) | (pad_amt << 25));
  g1[1] = (int)((td0 & 0xffffu) << 16);                   // tensor_dim0 lo
  g1[2] = (int)((td0 >> 16) | ((td1 & 0xffffu) << 16));   // td0 hi | td1 lo
  g1[3] = (int)((td1 >> 16) | (tile0 << 16));             // td1 hi | tile_dim0
  g1[4] = (int)(tile1 | (tile2 << 16));                   // tile_dim1 | tile_dim2
  g1[5] = (int)(u32)(s0 & 0xffffffffull);                 // tensor_dim0_stride lo32
  g1[6] = (int)((u32)((s0 >> 32) & 0xffffull) | ((u32)(s1 & 0xffffull) << 16));
  g1[7] = (int)(u32)((s1 >> 16) & 0xffffffffull);         // tensor_dim1_stride hi32
  v4i g2; g2[0] = (int)td2; g2[1] = 0; g2[2] = 0; g2[3] = 0;
  v4i g3 = {0, 0, 0, 0};
#if __clang_major__ >= 23
  v8i g4 = {0, 0, 0, 0, 0, 0, 0, 0};
  __builtin_amdgcn_tensor_load_to_lds(g0, g1, g2, g3, g4, 0);
#else
  __builtin_amdgcn_tensor_load_to_lds(g0, g1, g2, g3, 0);
#endif
}

// ---------------- prep: bf16-cast / pad the three weight matrices ----------------
__global__ __launch_bounds__(256) void esn_prep(
    const float* __restrict__ Win, const float* __restrict__ W,
    const float* __restrict__ Wout,
    u16* __restrict__ wb, u16* __restrict__ winp, u16* __restrict__ woutp)
{
  int idx = blockIdx.x * 256 + threadIdx.x;
  if (idx < RR * RR) wb[idx] = f2bf(W[idx]);               // row n, col k (B source)
  if (idx < RR * 32) {                                      // Win padded K 25 -> 32
    int n = idx >> 5, k = idx & 31;
    winp[idx] = f2bf(k < II ? Win[n * II + k] : 0.0f);
  }
  if (idx < 64 * RR) {                                      // Wout padded O 50 -> 64
    int o = idx >> 9, k = idx & 511;
    woutp[idx] = f2bf(o < OO ? Wout[o * RR + k] : 0.0f);
  }
}

// ---------------- persistent ESN kernel: 4 workgroups x 8 waves ----------------
__global__ __launch_bounds__(256, 1) void esn_main(
    const float* __restrict__ x, const u16* __restrict__ Wb,
    const u16* __restrict__ WinP, const u16* __restrict__ WoutP,
    float* __restrict__ out)
{
  __shared__ u16   hHist[TC * 16 * HST];   // bf16 h for the last TC steps (A source)
  __shared__ u16   woutLds[64 * HST];      // Wout B source, TDM-padded rows
  __shared__ float xStage[16 * TC * II];   // raw f32 x chunk (TDM 3D copy)
  __shared__ u16   xA[TC * 16 * 40];       // bf16 x A source, K padded to 32

  const int tid  = threadIdx.x;
  const int w    = tid >> 5;
  const int lane = tid & 31;
  const int half = lane >> 4;
  const int l16  = lane & 15;
  const int bg   = blockIdx.x;             // batch group: rows bg*16 .. bg*16+15

  // h(-1) = 0 lives in chunk slot TC-1
  for (int i = tid; i < 16 * HST; i += 256)
    hHist[(TC - 1) * 16 * HST + i] = 0;

  // TDM: 64KB Wout -> LDS, inserting 4 DWORDs of pad every 256 DWORDs
  // (row stride 1024B -> 1040B == HST elems)
  if (w == 0) {
    tdm_load_to_lds((u64)(const void*)WoutP, (u32)(u64)(const void*)&woutLds[0],
                    3u /*8B*/, 8192u, 0u, 8192u, 0u, 0u, 8192ull, 0ull, 0u,
                    1u /*pad*/, 7u /*256 DW*/, 3u /*4 DW*/);
    __builtin_amdgcn_s_wait_tensorcnt(0);
  }

  // Win B-fragments pinned in VGPRs (single K=32 tile covers padded I)
  v16bf winF[NTW];
#pragma unroll
  for (int j = 0; j < NTW; ++j) {
    int n = (w * NTW + j) * 16 + l16;
    winF[j] = *(const v16bf*)(WinP + n * 32 + half * 16);
  }

  // persistent f32 h state in C-layout: m = v + 8*half, n = tileBase + l16
  v8f hold[NTW];
#pragma unroll
  for (int j = 0; j < NTW; ++j)
#pragma unroll
    for (int v = 0; v < 8; ++v) hold[j][v] = 0.0f;

  // per-lane W bases: B-frag = 32 contiguous bytes of row n at col kt*32+half*16
  const u16* bBase[NTW];
#pragma unroll
  for (int j = 0; j < NTW; ++j)
    bBase[j] = Wb + (size_t)((w * NTW + j) * 16 + l16) * RR + half * 16;

  __syncthreads();

  for (int t0 = 0; t0 < TT; t0 += TC) {
    // ---- stage x[bg, t0..t0+TC, :] via 3D TDM descriptor ----
    if (w == 0) {
      u64 ga = (u64)(const void*)(x + ((size_t)bg * 16 * TT + (size_t)t0) * II);
      tdm_load_to_lds(ga, (u32)(u64)(const void*)&xStage[0],
                      2u /*4B*/, (u32)II, (u32)TC, (u32)II, (u32)TC, 16u,
                      (u64)II /*t stride*/, (u64)TT * II /*b stride*/, 16u,
                      0u, 0u, 0u);
      __builtin_amdgcn_s_wait_tensorcnt(0);
    }
    __syncthreads();

    // ---- build bf16 A-source for x, zero-padding K 25 -> 32 ----
    for (int i = tid; i < TC * 16 * 32; i += 256) {
      int tp = i >> 9, rem = i & 511, m = rem >> 5, k = rem & 31;
      float v = (k < II) ? xStage[(m * TC + tp) * II + k] : 0.0f;
      xA[tp * (16 * 40) + m * 40 + k] = f2bf(v);
    }
    __syncthreads();

    // ---- the recurrence: one LDS barrier per step ----
    // acc lives only within one step (32 VGPRs), seeded by the xin WMMA.
#pragma unroll
    for (int tp = 0; tp < TC; ++tp) {
      v8f acc[NTW];
      {
        v16bf ax = load_a_frag(xA + tp * (16 * 40) + l16 * 40 + half * 16);
        v8f z;
#pragma unroll
        for (int v = 0; v < 8; ++v) z[v] = 0.0f;
#pragma unroll
        for (int j = 0; j < NTW; ++j)
          acc[j] = wmma_bf16(ax, winF[j], z);      // acc = xin(t)
      }

      const u16* hPrev =
          hHist + ((tp + TC - 1) % TC) * (16 * HST) + l16 * HST + half * 16;
#pragma unroll 8
      for (int kt = 0; kt < 16; ++kt) {
        v16bf a = load_a_frag(hPrev + kt * 32);
#pragma unroll
        for (int j = 0; j < NTW; ++j) {            // 4 independent WMMA chains
          v16bf b = *(const v16bf*)(bBase[j] + kt * 32);
          acc[j] = wmma_bf16(a, b, acc[j]);
        }
      }

      // h = (1-a)*h + a*tanh(u); publish bf16 h to this step's history slot
      u16* hw = hHist + tp * (16 * HST);
#pragma unroll
      for (int j = 0; j < NTW; ++j) {
        int col = (w * NTW + j) * 16 + l16;
#pragma unroll
        for (int v = 0; v < 8; ++v) {
          float hn = (1.0f - ALPHA) * hold[j][v] + ALPHA * TANHF(acc[j][v]);
          hold[j][v] = hn;
          hw[(v + 8 * half) * HST + col] = f2bf(hn);
        }
      }
      __syncthreads();
    }

    // ---- readout for the whole chunk: 16 (tile,t) combos over 8 waves ----
#pragma unroll
    for (int c = 0; c < 2; ++c) {
      int combo = w * 2 + c;
      int tile = combo & 3, tp = combo >> 2;
      v8f oacc;
#pragma unroll
      for (int v = 0; v < 8; ++v) oacc[v] = 0.0f;
      const u16* ha = hHist + tp * (16 * HST) + l16 * HST + half * 16;
      const u16* wo = woutLds + (tile * 16 + l16) * HST + half * 16;
#pragma unroll 8
      for (int kt = 0; kt < 16; ++kt) {
        v16bf a = load_a_frag(ha + kt * 32);
        v16bf b = load_b_frag16(wo + kt * 32);
        oacc = wmma_bf16(a, b, oacc);
      }
      int o = tile * 16 + l16;
      if (o < OO) {
        size_t base = ((size_t)(bg * 16 + 8 * half) * TT + (size_t)(t0 + tp)) * OO + o;
#pragma unroll
        for (int v = 0; v < 8; ++v)
          out[base + (size_t)v * TT * OO] = oacc[v];
      }
    }
    // next iteration's xA-build barrier orders these reads before hHist reuse
  }
}

extern "C" void kernel_launch(void* const* d_in, const int* in_sizes, int n_in,
                              void* d_out, int out_size, void* d_ws, size_t ws_size,
                              hipStream_t stream) {
  (void)in_sizes; (void)n_in; (void)out_size; (void)ws_size;
  const float* x    = (const float*)d_in[0];
  const float* Win  = (const float*)d_in[1];
  const float* W    = (const float*)d_in[2];
  const float* Wout = (const float*)d_in[3];

  u16* wb    = (u16*)d_ws;          // 512x512 bf16      (512 KB)
  u16* winp  = wb + RR * RR;        // 512x32  bf16      ( 32 KB)
  u16* woutp = winp + RR * 32;      // 64x512  bf16      ( 64 KB)

  esn_prep<<<dim3((RR * RR + 255) / 256), dim3(256), 0, stream>>>(
      Win, W, Wout, wb, winp, woutp);
  esn_main<<<dim3(BB / 16), dim3(256), 0, stream>>>(
      x, wb, winp, woutp, (float*)d_out);
}
